// Net_18485539242413
// MI455X (gfx1250) — matrix operand order repaired and checked by reading, not compiled, not run on hardware
//
#include <hip/hip_runtime.h>
#include <hip/hip_bf16.h>

// ---------------------------------------------------------------------------
// Problem constants (from the reference)
// ---------------------------------------------------------------------------
#define BATCH        2048
#define IN_DIM       256
#define NUM_TREES    1024
#define DEPTH        7
#define LEAVES       128
#define INNER        127
#define NUM_FEATURES 26009
#define EXTRA        10
#define EPS_BN       1e-5f

// Feature dim padded to a multiple of the GEMM N tile (128): 204 * 128
#define FPAD         26112

// GEMM tiling
#define BM   64
#define BN   128
#define BK   32
#define LDA  80            // bytes per LDS row for A tile (32 bf16 = 64B + 16B pad)
#define LDB  80            // bytes per LDS row for B tile
#define ASZ  (BM * LDA)    // 5120 B
#define BSZ  (BN * LDB)    // 10240 B
#define BUFSZ (ASZ + BSZ)  // 15360 B per pipeline stage
#define KTILES (IN_DIM / BK)

// Workspace layout (bytes).  f kept in bf16 so the whole 107 MB intermediate
// stays resident in MI455X's 192 MB L2 across GEMM -> stats -> forest gathers.
#define OFF_WB    ((size_t)0)                                   // FPAD*256 bf16
#define OFF_XB    (OFF_WB + (size_t)FPAD * IN_DIM * 2)          // 2048*256 bf16
#define OFF_MEAN  (OFF_XB + (size_t)BATCH * IN_DIM * 2)         // FPAD f32
#define OFF_ISTD  (OFF_MEAN + (size_t)FPAD * 4)                 // FPAD f32
#define OFF_F     (OFF_ISTD + (size_t)FPAD * 4)                 // 2048*FPAD bf16

typedef __attribute__((ext_vector_type(16))) __bf16        v16bf;
typedef __attribute__((ext_vector_type(8)))  float         v8f;
typedef __attribute__((ext_vector_type(4)))  unsigned int  u32x4;

union FragCast {
    u32x4 u[2];
    v16bf v;
};

__device__ __forceinline__ unsigned short f32_to_bf16_rne(float f)
{
    union { float f; unsigned int u; } cv;
    cv.f = f;
    unsigned int u = cv.u;
    u += 0x7FFFu + ((u >> 16) & 1u);
    return (unsigned short)(u >> 16);
}

__device__ __forceinline__ float bf16_bits_to_f32(unsigned short h)
{
    union { unsigned int u; float f; } cv;
    cv.u = ((unsigned int)h) << 16;
    return cv.f;
}

// ---------------------------------------------------------------------------
// CDNA5 async global->LDS copy (ASYNCcnt-tracked), §15.18 opcode 98.
//   vdst VGPR = LDS byte offset (addr[31:0] of the shared pointer),
//   vaddr     = 64-bit global address pair, saddr = off.
// ---------------------------------------------------------------------------
__device__ __forceinline__ void async_copy_b128(unsigned lds_off, const void* gptr)
{
    unsigned long long ga = (unsigned long long)(uintptr_t)gptr;
    asm volatile("global_load_async_to_lds_b128 %0, %1, off"
                 :: "v"(lds_off), "v"(ga)
                 : "memory");
}

__device__ __forceinline__ void wait_async_le3()
{
    asm volatile("s_wait_asynccnt 0x3" ::: "memory");
}

__device__ __forceinline__ void wait_async_zero()
{
    asm volatile("s_wait_asynccnt 0x0" ::: "memory");
}

// ---------------------------------------------------------------------------
// Kernel 1: f32 -> bf16 (RNE), zero-fill padded tail
// ---------------------------------------------------------------------------
__global__ void cvt_bf16_kernel(const float* __restrict__ src,
                                unsigned short* __restrict__ dst,
                                long n_src, long n_total)
{
    long i = (long)blockIdx.x * blockDim.x + threadIdx.x;
    if (i >= n_total) return;
    unsigned short h = 0;
    if (i < n_src) h = f32_to_bf16_rne(src[i]);
    dst[i] = h;
}

// ---------------------------------------------------------------------------
// Kernel 2: WMMA GEMM  f[2048][FPAD] = x[2048][256] * W^T[256][FPAD]
//   grid = (FPAD/BN, BATCH/BM), block = 256 threads = 8 waves (wave32)
//   wave tile 32x32 = 2x2 v_wmma_f32_16x16x32_bf16 accumulators
//   K tiles staged through double-buffered LDS via async-to-LDS copies
//   Result stored as bf16 to keep the intermediate L2-resident.
// ---------------------------------------------------------------------------
__global__ void __launch_bounds__(256)
gemm_bf16_wmma_kernel(const unsigned short* __restrict__ xb,   // [BATCH][256] bf16
                      const unsigned short* __restrict__ wb,   // [FPAD][256] bf16
                      unsigned short* __restrict__ fbuf)       // [BATCH][FPAD] bf16
{
    __shared__ __align__(16) unsigned char lds[2 * BUFSZ];     // 30720 B

    const int tid   = threadIdx.x;
    const int lane  = tid & 31;
    const int wave  = tid >> 5;                 // 0..7
    const int waveM = (wave >> 2) * 32;         // 0 or 32
    const int waveN = (wave & 3) * 32;          // 0,32,64,96

    const int gM0 = blockIdx.y * BM;            // batch-row base
    const int gN0 = blockIdx.x * BN;            // feature-col base

    // LDS base byte offset of this block's allocation (low 32 bits of the shared address)
    const unsigned ldsBase = (unsigned)(uintptr_t)(&lds[0]);

    v8f acc[2][2];
    for (int mi = 0; mi < 2; ++mi)
        for (int ni = 0; ni < 2; ++ni)
            for (int e = 0; e < 8; ++e)
                acc[mi][ni][e] = 0.0f;

    // Per-thread staging assignments
    const int aRow = tid >> 2;                  // 0..63
    const int aKc  = (tid & 3) * 8;             // element offset within BK
    const int bRow = tid >> 1;                  // 0..127
    const int bKc  = (tid & 1) * 16;            // element offset within BK

    const unsigned short* aSrcBase = xb + (size_t)(gM0 + aRow) * IN_DIM + aKc;
    const unsigned short* bSrcBase = wb + (size_t)(gN0 + bRow) * IN_DIM + bKc;
    const unsigned aDstOff = (unsigned)(aRow * LDA + aKc * 2);
    const unsigned bDstOff = (unsigned)(ASZ + bRow * LDB + bKc * 2);

    // Prologue: asynchronously stage K-tile 0 into pipeline buffer 0
    {
        async_copy_b128(ldsBase + aDstOff,      aSrcBase);
        async_copy_b128(ldsBase + bDstOff,      bSrcBase);
        async_copy_b128(ldsBase + bDstOff + 16, bSrcBase + 8);
    }

#pragma unroll
    for (int it = 0; it < KTILES; ++it) {
        const int cur = it & 1;

        if (it + 1 < KTILES) {
            // Issue async staging for K-tile it+1 into the other buffer.
            // That buffer was last *read* in iteration it-1; the trailing
            // barrier of that iteration makes this write safe.
            const int ksn = (it + 1) * BK;
            const unsigned nb = (unsigned)(((it + 1) & 1) * BUFSZ);
            async_copy_b128(ldsBase + nb + aDstOff,      aSrcBase + ksn);
            async_copy_b128(ldsBase + nb + bDstOff,      bSrcBase + ksn);
            async_copy_b128(ldsBase + nb + bDstOff + 16, bSrcBase + ksn + 8);

            // Distant prefetch: two tiles ahead into the cache hierarchy
            if (it + 2 < KTILES) {
                __builtin_prefetch(aSrcBase + (it + 2) * BK, 0, 1);
                __builtin_prefetch(bSrcBase + (it + 2) * BK, 0, 1);
            }
            wait_async_le3();   // tile `it` complete; 3 newest (tile it+1) in flight
        } else {
            wait_async_zero();  // final tile complete
        }
        __syncthreads();        // tile `it` visible to all waves

        const unsigned char* ldsA = lds + cur * BUFSZ;
        const unsigned char* ldsB = lds + cur * BUFSZ + ASZ;

        // A fragments: 16-bit 16x32 layout.
        // lane<16 : row = base+lane,     K chunks {0..7, 16..23}
        // lane>=16: row = base+lane-16,  K chunks {8..15, 24..31}
        v16bf afrag[2];
        for (int mi = 0; mi < 2; ++mi) {
            int row = waveM + mi * 16 + (lane & 15);
            int kb  = (lane >> 4) * 16;             // byte offset of first chunk
            const unsigned char* pa = ldsA + row * LDA + kb;
            FragCast fc;
            fc.u[0] = *(const u32x4*)(pa);
            fc.u[1] = *(const u32x4*)(pa + 32);
            afrag[mi] = fc.v;
        }

        // B fragments: 16-bit 32x16 layout (column per lane, K chunks as A)
        v16bf bfrag[2];
        for (int ni = 0; ni < 2; ++ni) {
            int col = waveN + ni * 16 + (lane & 15);
            int kh  = (lane >> 4) * 16;             // byte offset of first chunk
            const unsigned char* pb = ldsB + col * LDB + kh;
            FragCast fc;
            fc.u[0] = *(const u32x4*)(pb);
            fc.u[1] = *(const u32x4*)(pb + 32);
            bfrag[ni] = fc.v;
        }

        // 2x2 WMMA
        for (int mi = 0; mi < 2; ++mi)
            for (int ni = 0; ni < 2; ++ni)
                acc[mi][ni] = __builtin_amdgcn_wmma_f32_16x16x32_bf16(
                    /*neg_a=*/false, afrag[mi],
                    /*neg_b=*/false, bfrag[ni],
                    /*c_mod=*/(short)0, acc[mi][ni],
                    /*reuse_a=*/false, /*reuse_b=*/false);

        __syncthreads();        // all waves done reading buffer `cur`
    }

    // Write-out (bf16): C/D 16x16 f32 layout —
    //   VGPR r: lanes 0-15 => M=r (N=lane), lanes 16-31 => M=r+8 (N=lane-16)
    for (int mi = 0; mi < 2; ++mi) {
        for (int ni = 0; ni < 2; ++ni) {
            int rowBase = gM0 + waveM + mi * 16 + 8 * (lane >> 4);
            int col     = gN0 + waveN + ni * 16 + (lane & 15);
            for (int r = 0; r < 8; ++r) {
                fbuf[(size_t)(rowBase + r) * FPAD + col] = f32_to_bf16_rne(acc[mi][ni][r]);
            }
        }
    }
}

// ---------------------------------------------------------------------------
// Kernel 3: per-feature batch statistics (mean, 1/sqrt(var+eps))
//   one thread per feature column, coalesced row sweep over bf16 f
// ---------------------------------------------------------------------------
__global__ void __launch_bounds__(256)
bn_stats_kernel(const unsigned short* __restrict__ fbuf,
                float* __restrict__ mean,
                float* __restrict__ istd)
{
    int c = blockIdx.x * blockDim.x + threadIdx.x;
    if (c >= FPAD) return;
    float s = 0.0f, sq = 0.0f;
    for (int r = 0; r < BATCH; ++r) {
        float v = bf16_bits_to_f32(fbuf[(size_t)r * FPAD + c]);
        s  += v;
        sq += v * v;
    }
    float m  = s * (1.0f / BATCH);
    float vv = sq * (1.0f / BATCH) - m * m;    // biased variance
    mean[c] = m;
    istd[c] = rsqrtf(vv + EPS_BN);
}

// ---------------------------------------------------------------------------
// Kernel 4: tree routing + leaf-weighted reduction
//   grid = BATCH blocks, 256 threads, 4 trees per thread
//   Random per-row gathers hit the L2-resident bf16 f matrix.
// ---------------------------------------------------------------------------
__global__ void __launch_bounds__(256)
forest_kernel(const unsigned short* __restrict__ fbuf,  // [BATCH][FPAD] bf16
              const float* __restrict__ mean,           // [FPAD]
              const float* __restrict__ istd,           // [FPAD]
              const float* __restrict__ thresholds,     // [T][INNER]
              const int*   __restrict__ ordinals,       // [T][INNER]
              const float* __restrict__ weights,        // [T][LEAVES][EXTRA]
              float* __restrict__ out)                  // [BATCH][EXTRA]
{
    __shared__ float sacc[EXTRA];
    const int b   = blockIdx.x;
    const int tid = threadIdx.x;

    if (tid < EXTRA) sacc[tid] = 0.0f;
    __syncthreads();

    const unsigned short* hrow = fbuf + (size_t)b * FPAD;

    float acc[EXTRA];
#pragma unroll
    for (int e = 0; e < EXTRA; ++e) acc[e] = 0.0f;

    for (int tt = 0; tt < NUM_TREES / 256; ++tt) {
        int t = tid + tt * 256;
        const int*   ord = ordinals   + (size_t)t * INNER;
        const float* thr = thresholds + (size_t)t * INNER;

        int   node   = 0;
        float margin = __builtin_inff();
#pragma unroll
        for (int d = 0; d < DEPTH; ++d) {
            int   feat = ord[node];
            float h    = (bf16_bits_to_f32(hrow[feat]) - mean[feat]) * istd[feat];
            float val  = h - thr[node];
            float a    = fabsf(val);
            margin = fminf(margin, a);
            node   = 2 * node + 1 + (val > 0.0f ? 1 : 0);
        }
        int leaf = node - INNER;
        const float* wl = weights + ((size_t)t * LEAVES + leaf) * EXTRA;
#pragma unroll
        for (int e = 0; e < EXTRA; ++e) acc[e] += margin * wl[e];
    }

#pragma unroll
    for (int e = 0; e < EXTRA; ++e) atomicAdd(&sacc[e], acc[e]);   // ds_add_f32
    __syncthreads();

    if (tid < EXTRA) out[(size_t)b * EXTRA + tid] = sacc[tid];
}

// ---------------------------------------------------------------------------
// Host-side launcher
// ---------------------------------------------------------------------------
extern "C" void kernel_launch(void* const* d_in, const int* in_sizes, int n_in,
                              void* d_out, int out_size, void* d_ws, size_t ws_size,
                              hipStream_t stream)
{
    const float* x          = (const float*)d_in[0];   // [2048, 256]
    const float* W          = (const float*)d_in[1];   // [26009, 256]
    const float* thresholds = (const float*)d_in[2];   // [1024, 127]
    const float* weights    = (const float*)d_in[3];   // [1024, 128, 10]
    const int*   ordinals   = (const int*)d_in[4];     // [1024, 127]
    float*       out        = (float*)d_out;           // [2048, 10]

    char* ws = (char*)d_ws;
    unsigned short* wb   = (unsigned short*)(ws + OFF_WB);
    unsigned short* xb   = (unsigned short*)(ws + OFF_XB);
    float*          mean = (float*)(ws + OFF_MEAN);
    float*          istd = (float*)(ws + OFF_ISTD);
    unsigned short* fbuf = (unsigned short*)(ws + OFF_F);

    // 1) bf16 conversions (W padded to FPAD rows with zeros)
    {
        long nW  = (long)NUM_FEATURES * IN_DIM;
        long nWt = (long)FPAD * IN_DIM;
        cvt_bf16_kernel<<<(int)((nWt + 255) / 256), 256, 0, stream>>>(W, wb, nW, nWt);

        long nX = (long)BATCH * IN_DIM;
        cvt_bf16_kernel<<<(int)((nX + 255) / 256), 256, 0, stream>>>(x, xb, nX, nX);
    }

    // 2) WMMA GEMM: f = x @ W^T   (raw, un-normalized, bf16 output)
    {
        dim3 grid(FPAD / BN, BATCH / BM);   // (204, 32)
        gemm_bf16_wmma_kernel<<<grid, 256, 0, stream>>>(xb, wb, fbuf);
    }

    // 3) BatchNorm statistics
    bn_stats_kernel<<<(FPAD + 255) / 256, 256, 0, stream>>>(fbuf, mean, istd);

    // 4) Forest traversal + output reduction
    forest_kernel<<<BATCH, 256, 0, stream>>>(fbuf, mean, istd, thresholds,
                                             ordinals, weights, out);
}
